// NodeUpdate_53730040873194
// MI455X (gfx1250) — matrix operand order repaired and checked by reading, not compiled, not run on hardware
//
#include <hip/hip_runtime.h>
#include <hip/hip_bf16.h>

// Problem constants (from reference)
#define NN 50000      // nodes
#define NE 500000     // edges
#define ND 256        // node feature dim
#define ED 128        // edge feature dim
#define GD 64         // global feature dim
// Derived
#define K1 (ND + ED)        // 384, edge MLP input
#define H1 (4 * ED)         // 512, edge MLP hidden
#define K3 (ND + ED + GD)   // 448, node MLP input
#define H3 (4 * ND)         // 1024, node MLP hidden

// LDS layout (bytes), edge kernel:
//   H (4 waves x 16x512 f16) | B buffer (2-wide slices, double buffered GEMM1)
#define E_LDS_H     0u
#define E_LDS_B     65536u
#define E_B2_STRIDE 24576u                       // 2 tiles x 12KB (GEMM1)
#define E_LDS_TOTAL (65536u + 2u * E_B2_STRIDE)  // 114,688 B -> 2 blocks/WGP
// LDS layout (bytes), node kernel:
//   H (2 waves x 16x1024 f16) | Y | ST | B region (32KB; GEMM1 dbl-buf inside)
#define N_LDS_H     0u
#define N_LDS_Y     65536u
#define N_LDS_ST    98304u
#define N_LDS_B     98560u
#define N_B1_STRIDE 16384u                       // GEMM1 slice stride (14KB used)
#define N_LDS_TOTAL (98560u + 32768u)            // 131,328 B -> 2 blocks/WGP

typedef __attribute__((ext_vector_type(16))) _Float16 v16h;
typedef __attribute__((ext_vector_type(8)))  _Float16 v8h;
typedef __attribute__((ext_vector_type(8)))  float    v8f;
typedef __attribute__((ext_vector_type(4)))  float    v4f;

static __device__ __forceinline__ v8f wmma_f16(v16h a, v16h b, v8f c) {
  // D = A(16x32 f16) * B(32x16 f16) + C(16x16 f32)
  return __builtin_amdgcn_wmma_f32_16x16x32_f16(
      false, a, false, b, (short)0, c, false, false);
}

// --- gfx1250 async global->LDS staging --------------------------------------
// 16 bytes per issue; tracked with ASYNCcnt, drained with s_wait_asynccnt.
static __device__ __forceinline__ void async_load_b128(unsigned lds_off,
                                                       const char* g) {
  asm volatile("global_load_async_to_lds_b128 %0, %1, off"
               :: "v"(lds_off), "v"(g) : "memory");
}
template <int N>
static __device__ __forceinline__ void wait_async() {
  asm volatile("s_wait_asynccnt %0" :: "n"(N) : "memory");
}
// Stage one packed-B column slice (all KT fragments of column tile ntt) into
// LDS. Fragments are 1024B each; packed base layout is [(kt*NT + nt)][1024B].
// KT*64 16B-chunks striped over T threads (exact division for all call sites).
template <int KT, int T>
static __device__ __forceinline__ void stage_slice(const char* wbase, int NT,
                                                   int ntt, unsigned ldsBase,
                                                   int tid) {
  for (int ci = tid; ci < KT * 64; ci += T) {
    const int kt = ci >> 6, rem = ci & 63;
    async_load_b128(ldsBase + ((unsigned)ci << 4),
                    wbase + (((size_t)(kt * NT + ntt)) << 10) + ((size_t)rem << 4));
  }
}
// Two consecutive column tiles, packed back-to-back in LDS.
template <int KT, int T>
static __device__ __forceinline__ void stage_slice2(const char* wbase, int NT,
                                                    int ntt, unsigned ldsBase,
                                                    int tid) {
  stage_slice<KT, T>(wbase, NT, ntt,     ldsBase,                        tid);
  stage_slice<KT, T>(wbase, NT, ntt + 1, ldsBase + (unsigned)KT * 1024u, tid);
}

// LDS fragment loads feeding WMMA (wave32 layouts per CDNA5 ISA tables).
static __device__ __forceinline__ v16h ldb_frag(const _Float16* bp, int kt,
                                                int lane) {
  return *(const v16h*)(bp + kt * 512 + lane * 16);     // 2x ds_load_b128
}
static __device__ __forceinline__ v16h lda_frag(const _Float16* hrow, int kt,
                                                int hi) {
  const int kc = kt * 32 + hi * 8;
  v8h lo = *(const v8h*)(hrow + kc);
  v8h hh = *(const v8h*)(hrow + kc + 16);
  return __builtin_shufflevector(lo, hh, 0,1,2,3,4,5,6,7,8,9,10,11,12,13,14,15);
}

// ---------------------------------------------------------------------------
// Pack a row-major fp32 weight [K,N] into f16 WMMA-B fragment order:
// lane holds column n = nt*16 + (lane&15); k = kt*32 + (lane>>4)*8 + (j&7) + (j>>3)*16.
// ---------------------------------------------------------------------------
__global__ void gn_pack_b_f16(const float* __restrict__ src,
                              _Float16* __restrict__ dst, int K, int N) {
  int o = blockIdx.x * blockDim.x + threadIdx.x;
  if (o >= K * N) return;
  int NT = N >> 4;
  int tile   = o >> 9;       // 512 elements per 32x16 tile
  int within = o & 511;
  int lane = within >> 4;
  int j    = within & 15;
  int kt = tile / NT;
  int nt = tile - kt * NT;
  int k = kt * 32 + ((lane >> 4) << 3) + (j & 7) + ((j >> 3) << 4);
  int n = (nt << 4) + (lane & 15);
  dst[o] = (_Float16)src[(size_t)k * N + n];
}

// degree count (scatter-mean denominator)
__global__ void gn_count(const int* __restrict__ col, float* __restrict__ cnt) {
  int e = blockIdx.x * blockDim.x + threadIdx.x;
  if (e < NE) atomicAdd(&cnt[col[e]], 1.0f);
}

// ---------------------------------------------------------------------------
// Edge MLP: e = relu([x[row] || edge_attr] @ W1 + b1) @ W2 + b2, then atomic
// scatter-add into s[col]. Block = 128 threads (4 waves), 64 edges per block.
// Weights async-staged to LDS; two N-tiles per iteration share A fragments,
// giving two independent WMMA chains per DS wait.
// ---------------------------------------------------------------------------
__global__ void gn_edge_mlp(const float* __restrict__ x,
                            const float* __restrict__ eattr,
                            const _Float16* __restrict__ W1p, const float* __restrict__ b1,
                            const _Float16* __restrict__ W2p, const float* __restrict__ b2,
                            const int* __restrict__ row, const int* __restrict__ col,
                            float* __restrict__ s) {
  extern __shared__ char esm[];
  const int tid  = threadIdx.x;
  const int lane = tid & 31;
  const int wave = tid >> 5;
  _Float16* H = (_Float16*)(esm + E_LDS_H) + wave * (16 * H1);
  const int m  = lane & 15;
  const int hi = lane >> 4;
  const int ebase = blockIdx.x * 64 + wave * 16;
  int e = ebase + m; if (e >= NE) e = NE - 1;
  const float* __restrict__ xr = x + (size_t)row[e] * ND;
  const float* __restrict__ ar = eattr + (size_t)e * ED;

  // Kick off tiles {0,1} of W1 before doing the gather work.
  stage_slice2<K1 / 32, 128>((const char*)W1p, H1 / 16, 0, E_LDS_B, tid);

  // Build all 12 A fragments (K=384) once; reuse across all N-tiles.
  v16h afrag[K1 / 32];
#pragma unroll
  for (int kt = 0; kt < K1 / 32; ++kt) {
    v16h a;
    const int kc = kt * 32 + hi * 8;
#pragma unroll
    for (int h2 = 0; h2 < 2; ++h2) {
      const int k = kc + h2 * 16;
      const float* p = (k < ND) ? (xr + k) : (ar + (k - ND));
      v4f f0 = ((const v4f*)p)[0];
      v4f f1 = ((const v4f*)p)[1];
#pragma unroll
      for (int j = 0; j < 4; ++j) {
        a[h2 * 8 + j]     = (_Float16)f0[j];
        a[h2 * 8 + 4 + j] = (_Float16)f1[j];
      }
    }
    afrag[kt] = a;
  }

  // GEMM1: [16,384] @ W1 [384,512] -> relu -> LDS; 2 column tiles / iter
  for (int nt = 0; nt < H1 / 16; nt += 2) {
    const unsigned cb = E_LDS_B + (unsigned)((nt >> 1) & 1) * E_B2_STRIDE;
    if (nt + 2 < H1 / 16) {
      stage_slice2<K1 / 32, 128>((const char*)W1p, H1 / 16, nt + 2,
          E_LDS_B + (unsigned)(((nt >> 1) + 1) & 1) * E_B2_STRIDE, tid);
      wait_async<2 * (K1 / 32) * 64 / 128>();   // current pair done
    } else {
      wait_async<0>();
    }
    __syncthreads();
    const _Float16* bp0 = (const _Float16*)(esm + cb);
    const _Float16* bp1 = bp0 + (K1 / 32) * 512;
    v8f acc0 = {}, acc1 = {};
#pragma unroll
    for (int kt = 0; kt < K1 / 32; ++kt) {
      v16h b0 = ldb_frag(bp0, kt, lane);
      v16h b1 = ldb_frag(bp1, kt, lane);
      acc0 = wmma_f16(afrag[kt], b0, acc0);
      acc1 = wmma_f16(afrag[kt], b1, acc1);
    }
    const float bias0 = b1[nt * 16 + m];
    const float bias1 = b1[(nt + 1) * 16 + m];
#pragma unroll
    for (int r = 0; r < 8; ++r) {
      float v0 = acc0[r] + bias0; v0 = v0 > 0.0f ? v0 : 0.0f;
      float v1 = acc1[r] + bias1; v1 = v1 > 0.0f ? v1 : 0.0f;
      _Float16* hr = H + (r + hi * 8) * H1 + nt * 16 + m;
      hr[0]  = (_Float16)v0;
      hr[16] = (_Float16)v1;
    }
    __syncthreads();   // protect B buffer before it is re-staged
  }

  // GEMM2: [16,512] @ W2 [512,128] -> +b2 -> atomic scatter-add to s[col]
  // Single-buffered 2-wide slices (32KB fits the 48KB region).
  const _Float16* hrow = H + m * H1;
  for (int nt2 = 0; nt2 < ED / 16; nt2 += 2) {
    stage_slice2<H1 / 32, 128>((const char*)W2p, ED / 16, nt2, E_LDS_B, tid);
    wait_async<0>();
    __syncthreads();
    const _Float16* bp0 = (const _Float16*)(esm + E_LDS_B);
    const _Float16* bp1 = bp0 + (H1 / 32) * 512;
    v8f acc0 = {}, acc1 = {};
#pragma unroll
    for (int kt = 0; kt < H1 / 32; ++kt) {
      v16h a  = lda_frag(hrow, kt, hi);
      acc0 = wmma_f16(a, ldb_frag(bp0, kt, lane), acc0);
      acc1 = wmma_f16(a, ldb_frag(bp1, kt, lane), acc1);
    }
    const float bias0 = b2[nt2 * 16 + m];
    const float bias1 = b2[(nt2 + 1) * 16 + m];
#pragma unroll
    for (int r = 0; r < 8; ++r) {
      const int ee = ebase + r + hi * 8;
      if (ee < NE) {
        const int c = col[ee];
        float* srow = &s[(size_t)c * ED + nt2 * 16 + m];
        atomicAdd(srow,      acc0[r] + bias0);
        atomicAdd(srow + 16, acc1[r] + bias1);
      }
    }
    __syncthreads();   // all reads done before buffer is re-staged
  }
}

// ---------------------------------------------------------------------------
// Node MLP: h = relu([x || s/max(cnt,1) || u[batch]] @ W3 + b3) @ W4 + b4
// y = h + x ; LayerNorm(y) * gamma + beta -> out
// Block = 64 threads (2 waves), 32 nodes per block. Async-staged weights.
// ---------------------------------------------------------------------------
__global__ void gn_node_mlp(const float* __restrict__ x,
                            const float* __restrict__ s,
                            const float* __restrict__ cnt,
                            const float* __restrict__ u,
                            const int* __restrict__ batch,
                            const _Float16* __restrict__ W3p, const float* __restrict__ b3,
                            const _Float16* __restrict__ W4p, const float* __restrict__ b4,
                            const float* __restrict__ gamma, const float* __restrict__ beta,
                            float* __restrict__ out) {
  extern __shared__ char smem[];
  const int tid  = threadIdx.x;
  const int lane = tid & 31;
  const int wave = tid >> 5;     // 0..1
  _Float16* H = (_Float16*)(smem + N_LDS_H) + wave * (16 * H3);
  float* Y  = (float*)(smem + N_LDS_Y) + wave * (16 * ND);
  float* ST = (float*)(smem + N_LDS_ST) + wave * 32;
  const int m  = lane & 15;
  const int hi = lane >> 4;
  const int nbase = blockIdx.x * 32 + wave * 16;
  int node = nbase + m; if (node >= NN) node = NN - 1;
  const float* __restrict__ xr = x + (size_t)node * ND;
  const float* __restrict__ sr = s + (size_t)node * ED;
  const float* __restrict__ ur = u + (size_t)batch[node] * GD;
  const float inv = 1.0f / fmaxf(cnt[node], 1.0f);

  stage_slice<K3 / 32, 64>((const char*)W3p, H3 / 16, 0, N_LDS_B, tid);

  // Build all 14 A fragments (K=448): [x(0:256) | agg(256:384) | u(384:448)]
  v16h afrag[K3 / 32];
#pragma unroll
  for (int kt = 0; kt < K3 / 32; ++kt) {
    v16h a;
    const int kc = kt * 32 + hi * 8;
#pragma unroll
    for (int h2 = 0; h2 < 2; ++h2) {
      const int k = kc + h2 * 16;
      v4f f0, f1;
      if (k < ND)           { const v4f* p = (const v4f*)(xr + k);             f0 = p[0];       f1 = p[1]; }
      else if (k < ND + ED) { const v4f* p = (const v4f*)(sr + (k - ND));      f0 = p[0] * inv; f1 = p[1] * inv; }
      else                  { const v4f* p = (const v4f*)(ur + (k - ND - ED)); f0 = p[0];       f1 = p[1]; }
#pragma unroll
      for (int j = 0; j < 4; ++j) {
        a[h2 * 8 + j]     = (_Float16)f0[j];
        a[h2 * 8 + 4 + j] = (_Float16)f1[j];
      }
    }
    afrag[kt] = a;
  }

  // GEMM1: [16,448] @ W3 [448,1024] -> relu -> LDS (double-buffered slices)
  for (int nt = 0; nt < H3 / 16; ++nt) {
    const unsigned cb = N_LDS_B + (unsigned)(nt & 1) * N_B1_STRIDE;
    if (nt + 1 < H3 / 16) {
      stage_slice<K3 / 32, 64>((const char*)W3p, H3 / 16, nt + 1,
                               N_LDS_B + (unsigned)((nt + 1) & 1) * N_B1_STRIDE, tid);
      wait_async<(K3 / 32) * 64 / 64>();
    } else {
      wait_async<0>();
    }
    __syncthreads();
    const _Float16* bp = (const _Float16*)(smem + cb);
    v8f acc = {};
#pragma unroll
    for (int kt = 0; kt < K3 / 32; ++kt) {
      acc = wmma_f16(afrag[kt], ldb_frag(bp, kt, lane), acc);
    }
    const float bias = b3[nt * 16 + m];
#pragma unroll
    for (int r = 0; r < 8; ++r) {
      float v = acc[r] + bias;
      v = v > 0.0f ? v : 0.0f;
      H[(r + hi * 8) * H3 + nt * 16 + m] = (_Float16)v;
    }
    __syncthreads();
  }

  // GEMM2: [16,1024] @ W4 [1024,256] -> + b4 + residual x -> Y (f32, LDS)
  // Single-buffered slices (32KB region).
  const _Float16* hrow = H + m * H3;
  for (int nt2 = 0; nt2 < ND / 16; ++nt2) {
    stage_slice<H3 / 32, 64>((const char*)W4p, ND / 16, nt2, N_LDS_B, tid);
    wait_async<0>();
    __syncthreads();
    const _Float16* bp = (const _Float16*)(smem + N_LDS_B);
    v8f acc = {};
#pragma unroll
    for (int kt = 0; kt < H3 / 32; ++kt) {
      acc = wmma_f16(lda_frag(hrow, kt, hi), ldb_frag(bp, kt, lane), acc);
    }
    const float bias = b4[nt2 * 16 + m];
#pragma unroll
    for (int r = 0; r < 8; ++r) {
      const int mr = r + hi * 8;
      int nd = nbase + mr; if (nd >= NN) nd = NN - 1;
      const float xv = x[(size_t)nd * ND + nt2 * 16 + m];
      Y[mr * ND + nt2 * 16 + m] = acc[r] + bias + xv;
    }
    __syncthreads();
  }

  // Per-row LayerNorm statistics (lanes 0..15, one row each)
  if (lane < 16) {
    float sum = 0.0f, sq = 0.0f;
    const float* yr = Y + lane * ND;
    for (int f = 0; f < ND; ++f) { float v = yr[f]; sum += v; sq += v * v; }
    const float mu = sum * (1.0f / (float)ND);
    float var = sq * (1.0f / (float)ND) - mu * mu;
    var = var > 0.0f ? var : 0.0f;
    ST[lane]      = mu;
    ST[16 + lane] = rsqrtf(var + 1e-5f);
  }
  __syncthreads();

  // Normalize + affine, write out
  for (int idx = lane; idx < 16 * ND; idx += 32) {
    const int r = idx >> 8, f = idx & (ND - 1);
    const int nd = nbase + r;
    if (nd < NN)
      out[(size_t)nd * ND + f] = (Y[idx] - ST[r]) * ST[16 + r] * gamma[f] + beta[f];
  }
}

// ---------------------------------------------------------------------------
extern "C" void kernel_launch(void* const* d_in, const int* in_sizes, int n_in,
                              void* d_out, int out_size, void* d_ws, size_t ws_size,
                              hipStream_t stream) {
  (void)in_sizes; (void)n_in; (void)out_size; (void)ws_size;
  const float* x     = (const float*)d_in[0];
  const float* eat   = (const float*)d_in[1];
  const float* u     = (const float*)d_in[2];
  const float* W1    = (const float*)d_in[3];
  const float* b1    = (const float*)d_in[4];
  const float* W2    = (const float*)d_in[5];
  const float* b2    = (const float*)d_in[6];
  const float* W3    = (const float*)d_in[7];
  const float* b3    = (const float*)d_in[8];
  const float* W4    = (const float*)d_in[9];
  const float* b4    = (const float*)d_in[10];
  const float* gamma = (const float*)d_in[11];
  const float* beta  = (const float*)d_in[12];
  const int* eidx    = (const int*)d_in[13];
  const int* batch   = (const int*)d_in[14];
  const int* row = eidx;
  const int* col = eidx + NE;
  float* out = (float*)d_out;

  // Workspace carve-up (all 256B-aligned)
  char* ws = (char*)d_ws;
  size_t off = 0;
  float* s = (float*)(ws + off);            off += (size_t)NN * ED * 4;              // 25,600,000
  float* cnt = (float*)(ws + off);          off += ((size_t)NN * 4 + 255) & ~255ull; // 200,192
  const size_t zero_bytes = off;
  _Float16* W1p = (_Float16*)(ws + off);    off += (size_t)K1 * H1 * 2;
  _Float16* W2p = (_Float16*)(ws + off);    off += (size_t)H1 * ED * 2;
  _Float16* W3p = (_Float16*)(ws + off);    off += (size_t)K3 * H3 * 2;
  _Float16* W4p = (_Float16*)(ws + off);    off += (size_t)H3 * ND * 2;

  // Allow large dynamic LDS (ignore errors; AMD usually permits device max)
  (void)hipFuncSetAttribute((const void*)gn_edge_mlp,
      hipFuncAttributeMaxDynamicSharedMemorySize, (int)E_LDS_TOTAL);
  (void)hipFuncSetAttribute((const void*)gn_node_mlp,
      hipFuncAttributeMaxDynamicSharedMemorySize, (int)N_LDS_TOTAL);

  // 1) zero segment-sum + counts
  hipMemsetAsync(ws, 0, zero_bytes, stream);

  // 2) pack weights into f16 WMMA-B fragment order
  gn_pack_b_f16<<<(K1 * H1 + 255) / 256, 256, 0, stream>>>(W1, W1p, K1, H1);
  gn_pack_b_f16<<<(H1 * ED + 255) / 256, 256, 0, stream>>>(W2, W2p, H1, ED);
  gn_pack_b_f16<<<(K3 * H3 + 255) / 256, 256, 0, stream>>>(W3, W3p, K3, H3);
  gn_pack_b_f16<<<(H3 * ND + 255) / 256, 256, 0, stream>>>(W4, W4p, H3, ND);

  // 3) in-degree counts
  gn_count<<<(NE + 255) / 256, 256, 0, stream>>>(col, cnt);

  // 4) edge MLP + scatter-add
  gn_edge_mlp<<<(NE + 63) / 64, 128, E_LDS_TOTAL, stream>>>(
      x, eat, W1p, b1, W2p, b2, row, col, s);

  // 5) node MLP + residual + layernorm
  gn_node_mlp<<<(NN + 31) / 32, 64, N_LDS_TOTAL, stream>>>(
      x, s, cnt, u, batch, W3p, b3, W4p, b4, gamma, beta, out);
}
